// GraphAttentionLayer_71605694758948
// MI455X (gfx1250) — matrix operand order repaired and checked by reading, not compiled, not run on hardware
//
#include <hip/hip_runtime.h>

#define N_NODES 8192
#define F_IN    256
#define F_OUT   128
#define NEG_BIG (-1e12f)
#define ALPHA   0.2f

typedef __attribute__((ext_vector_type(16))) _Float16 v16h;
typedef __attribute__((ext_vector_type(8)))  _Float16 v8h;
typedef __attribute__((ext_vector_type(8)))  float    v8f;

__device__ __forceinline__ v8f wmma_f16(v16h a, v16h b, v8f c) {
    return __builtin_amdgcn_wmma_f32_16x16x32_f16(false, a, false, b, (short)0, c,
                                                  false, false);
}

// A-fragment (16x32, f16): lane l holds row M=(l&15); K = k0 + (l>>4)*16 + {0..15}
// in halves 0..15.  Source: row-major f16, leading dim ld.
__device__ __forceinline__ v16h ld_a_f16(const _Float16* base, int ld, int row0,
                                         int k0, int lane) {
    int r = lane & 15, hi = lane >> 4;
    const _Float16* q = base + (size_t)(row0 + r) * ld + k0 + hi * 16;
    v8h lo = *(const v8h*)q;
    v8h hh = *(const v8h*)(q + 8);
    v16h v;
#pragma unroll
    for (int t = 0; t < 8; ++t) { v[t] = lo[t]; v[8 + t] = hh[t]; }
    return v;
}

// Same A-fragment but source is row-major f32 (convert on load).
__device__ __forceinline__ v16h ld_a_f32(const float* base, int ld, int row0,
                                         int k0, int lane) {
    int r = lane & 15, hi = lane >> 4;
    const float* q = base + (size_t)(row0 + r) * ld + k0 + hi * 16;
    v16h v;
#pragma unroll
    for (int t = 0; t < 8; ++t) {
        v[t]     = (_Float16)q[t];
        v[8 + t] = (_Float16)q[8 + t];
    }
    return v;
}

// B-fragment (32x16, f16): lane l holds column N=(l&15); K = k0 + (l>>4)*16 + t.
// Here B[k][n] = src[col0+n][k]  (B columns are rows of a row-major f16 matrix),
// so each lane does one contiguous 32-byte load.
__device__ __forceinline__ v16h ld_bT_f16(const _Float16* base, int ld, int col0,
                                          int k0, int lane) {
    int n = lane & 15, hi = lane >> 4;
    return *(const v16h*)(base + (size_t)(col0 + n) * ld + k0 + hi * 16);
}

// B-fragment from a true row-major f32 matrix B[k][n] (strided gather + convert).
// Only used for the tiny 256x128 weight matrices (L2 resident).
__device__ __forceinline__ v16h ld_b_f32(const float* base, int ld, int n0,
                                         int k0, int lane) {
    int n = lane & 15, hi = lane >> 4;
    const float* q = base + (size_t)(k0 + hi * 16) * ld + n0 + n;
    v16h v;
#pragma unroll
    for (int t = 0; t < 16; ++t) v[t] = (_Float16)q[(size_t)t * ld];
    return v;
}

// ---------------------------------------------------------------------------
// Stage 1: h = inp@W, h2 = inp@W2 (row-major f16), h3T = (inp@W3)^T (f16).
// One wave per 16-row tile; blockIdx.y selects which projection.
// ---------------------------------------------------------------------------
__global__ void gat_stage1(const float* __restrict__ inp,
                           const float* __restrict__ W0,
                           const float* __restrict__ W1,
                           const float* __restrict__ W2,
                           _Float16* __restrict__ h,
                           _Float16* __restrict__ h2,
                           _Float16* __restrict__ h3T) {
    int lane  = threadIdx.x & 31;
    int r_    = lane & 15, hi = lane >> 4;
    int i0    = blockIdx.x * 16;
    int which = blockIdx.y;
    const float* W = (which == 0) ? W0 : (which == 1 ? W1 : W2);

    // Hoist the inp A-fragments (K = 256 -> 8 chunks of 32).
    v16h A[8];
#pragma unroll
    for (int kc = 0; kc < 8; ++kc) A[kc] = ld_a_f32(inp, F_IN, i0, kc * 32, lane);

#pragma unroll 1
    for (int ft = 0; ft < 8; ++ft) {
        v8f c = {};
#pragma unroll
        for (int kc = 0; kc < 8; ++kc) {
            v16h B = ld_b_f32(W, F_OUT, ft * 16, kc * 32, lane);
            c = wmma_f16(A[kc], B, c);
        }
        // C layout: lane holds D[m = r + 8*hi][f = ft*16 + (lane&15)], r = 0..7
        if (which == 2) {
#pragma unroll
            for (int r = 0; r < 8; ++r)
                h3T[(size_t)(ft * 16 + r_) * N_NODES + i0 + r + 8 * hi] =
                    (_Float16)c[r];
        } else {
            _Float16* dst = (which == 0) ? h : h2;
#pragma unroll
            for (int r = 0; r < 8; ++r)
                dst[(size_t)(i0 + r + 8 * hi) * F_OUT + ft * 16 + r_] =
                    (_Float16)c[r];
        }
    }
}

// ---------------------------------------------------------------------------
// Stage 2: fused  e = leaky(h @ h2^T) -> mask(adj) -> online softmax -> @ h3
// One wave per 16-row tile; 8 waves per block (L1 reuse of h2/h3T/adj tiles).
// Score tiles are computed TRANSPOSED (A = h2 rows, B = h rows) so that each
// lane owns one attention row: softmax stats are per-lane + one shfl_xor(16).
// ---------------------------------------------------------------------------
__global__ void gat_stage2(const _Float16* __restrict__ h,
                           const _Float16* __restrict__ h2,
                           const _Float16* __restrict__ h3T,
                           const int* __restrict__ adj,
                           float* __restrict__ out) {
    int lane = threadIdx.x & 31;
    int wave = threadIdx.x >> 5;
    int i0   = (blockIdx.x * (blockDim.x >> 5) + wave) * 16;
    int r_   = lane & 15, hi = lane >> 4;

    // Loop-invariant B-fragments: rows i0..i0+15 of h along K=0..127.
    v16h Bh[4];
#pragma unroll
    for (int kc = 0; kc < 4; ++kc) Bh[kc] = ld_bT_f16(h, F_OUT, i0, kc * 32, lane);

    v8f acc[8] = {};          // acc[ft] holds h_prime^T tile [f][m]
    float m_run = NEG_BIG;    // running row max (row = lane&15)
    float l_run = 0.f;        // running row sum

#pragma unroll 1
    for (int j = 0; j < N_NODES; j += 32) {
        // --- score tiles (transposed): t0 = e^T[j..j+15][i0..], t1 = +16 ---
        v8f t0 = {}, t1 = {};
#pragma unroll
        for (int kc = 0; kc < 4; ++kc) {
            v16h a0 = ld_a_f16(h2, F_OUT, j, kc * 32, lane);
            t0 = wmma_f16(a0, Bh[kc], t0);
        }
#pragma unroll
        for (int kc = 0; kc < 4; ++kc) {
            v16h a1 = ld_a_f16(h2, F_OUT, j + 16, kc * 32, lane);
            t1 = wmma_f16(a1, Bh[kc], t1);
        }

        // --- adjacency: lane needs adj[i0 + (lane&15)][j + n] for its n's ---
        const int* arow = adj + (size_t)(i0 + r_) * N_NODES + j + 8 * hi;
        int4 x0 = *(const int4*)(arow);
        int4 x1 = *(const int4*)(arow + 4);
        int4 y0 = *(const int4*)(arow + 16);
        int4 y1 = *(const int4*)(arow + 20);
        int av0[8] = {x0.x, x0.y, x0.z, x0.w, x1.x, x1.y, x1.z, x1.w};
        int av1[8] = {y0.x, y0.y, y0.z, y0.w, y1.x, y1.y, y1.z, y1.w};

        // --- leaky relu + mask + per-row running max ---
        float s0[8], s1[8];
        float mloc = m_run;
#pragma unroll
        for (int r = 0; r < 8; ++r) {
            float e0 = t0[r]; e0 = (e0 > 0.f) ? e0 : ALPHA * e0;
            float e1 = t1[r]; e1 = (e1 > 0.f) ? e1 : ALPHA * e1;
            s0[r] = (av0[r] > 0) ? e0 : NEG_BIG;
            s1[r] = (av1[r] > 0) ? e1 : NEG_BIG;
            mloc = fmaxf(mloc, fmaxf(s0[r], s1[r]));
        }
        mloc = fmaxf(mloc, __shfl_xor(mloc, 16, 32));   // merge column halves
        float scale = __expf(m_run - mloc);

        float p0[8], p1[8], lsum = 0.f;
#pragma unroll
        for (int r = 0; r < 8; ++r) {
            p0[r] = (s0[r] > -1e11f) ? __expf(s0[r] - mloc) : 0.f;
            p1[r] = (s1[r] > -1e11f) ? __expf(s1[r] - mloc) : 0.f;
            lsum += p0[r] + p1[r];
        }
        lsum += __shfl_xor(lsum, 16, 32);
        l_run = l_run * scale + lsum;
        m_run = mloc;

        // Rescale accumulator (per-lane scalar = per attention row).
#pragma unroll
        for (int ft = 0; ft < 8; ++ft)
#pragma unroll
            for (int r = 0; r < 8; ++r) acc[ft][r] *= scale;

        // --- build P^T as a B-fragment: lane needs P[K = (hi*16)+t][m=l&15] ---
        v16h Pb;
#pragma unroll
        for (int t = 0; t < 8; ++t) {
            float sp0 = __shfl_xor(p0[t], 16, 32);
            float sp1 = __shfl_xor(p1[t], 16, 32);
            Pb[t]     = (_Float16)(hi ? sp1 : p0[t]);   // K = hi*16 + t
            Pb[8 + t] = (_Float16)(hi ? p1[t] : sp0);   // K = hi*16 + 8 + t
        }

        // --- acc[ft] += h3T(A) x P^T(B):  D[f][m] += sum_n h3[n][f] p[m][n] ---
#pragma unroll
        for (int ft = 0; ft < 8; ++ft) {
            v16h Av = ld_a_f16(h3T, N_NODES, ft * 16, j, lane);
            acc[ft] = wmma_f16(Av, Pb, acc[ft]);
        }
    }

    // --- finalize: divide by row sum, ELU, scatter h_prime^T -> row-major ---
    float inv = (l_run > 0.f) ? 1.f / l_run : 0.f;
#pragma unroll
    for (int ft = 0; ft < 8; ++ft)
#pragma unroll
        for (int r = 0; r < 8; ++r) {
            float v = acc[ft][r] * inv;
            v = (v > 0.f) ? v : (__expf(v) - 1.f);
            out[(size_t)(i0 + r_) * F_OUT + ft * 16 + 8 * hi + r] = v;
        }
}

extern "C" void kernel_launch(void* const* d_in, const int* in_sizes, int n_in,
                              void* d_out, int out_size, void* d_ws, size_t ws_size,
                              hipStream_t stream) {
    const float* inp = (const float*)d_in[0];
    const int*   adj = (const int*)d_in[1];
    const float* W   = (const float*)d_in[2];
    const float* W2  = (const float*)d_in[3];
    const float* W3  = (const float*)d_in[4];
    float* out = (float*)d_out;

    _Float16* hbuf  = (_Float16*)d_ws;                       // 2 MB
    _Float16* h2buf = hbuf + (size_t)N_NODES * F_OUT;        // 2 MB
    _Float16* h3T   = h2buf + (size_t)N_NODES * F_OUT;       // 2 MB (transposed)

    gat_stage1<<<dim3(N_NODES / 16, 3), 32, 0, stream>>>(inp, W, W2, W3,
                                                         hbuf, h2buf, h3T);
    gat_stage2<<<dim3(N_NODES / 16 / 8), 256, 0, stream>>>(hbuf, h2buf, h3T,
                                                           adj, out);
}
